// DynamicCapsules_40578851012825
// MI455X (gfx1250) — matrix operand order repaired and checked by reading, not compiled, not run on hardware
//
#include <hip/hip_runtime.h>
#include <hip/hip_bf16.h>

typedef __attribute__((ext_vector_type(16))) _Float16 v16h;
typedef __attribute__((ext_vector_type(8)))  _Float16 h8;
typedef __attribute__((ext_vector_type(8)))  float    v8f;

// ---------------------------------------------------------------------------
// fragment helpers (CDNA5 ISA 16-bit A/B VGPR layouts)
// ---------------------------------------------------------------------------
__device__ __forceinline__ v16h cat8(h8 lo, h8 hi) {
    return __builtin_shufflevector(lo, hi, 0,1,2,3,4,5,6,7,8,9,10,11,12,13,14,15);
}
__device__ __forceinline__ v16h load_A_frag(const _Float16* __restrict__ Arow, int hi) {
    h8 lo = *(const h8*)(Arow + (hi << 3));        // K = hi*8 .. +7
    h8 hv = *(const h8*)(Arow + 16 + (hi << 3));   // K = 16+hi*8 .. +7
    return cat8(lo, hv);
}
__device__ __forceinline__ v16h load_B_frag(const _Float16* brow, int hi) {
    h8 lo = *(const h8*)(brow + (hi << 3));        // K = hi*8 .. +7
    h8 hv = *(const h8*)(brow + 16 + (hi << 3));   // K = 16+hi*8 .. +7
    return cat8(lo, hv);
}

// ---------------------------------------------------------------------------
// conversion / table kernels
// ---------------------------------------------------------------------------
__global__ void cvt_f32_to_f16(const float* __restrict__ src,
                               _Float16* __restrict__ dst, int n) {
    int i = blockIdx.x * blockDim.x + threadIdx.x;
    if (i < n) dst[i] = (_Float16)src[i];
}

// conv1 weights [64][243] -> [64][256] f16, zero padded K
__global__ void build_A1(const float* __restrict__ w, _Float16* __restrict__ dst) {
    int i = blockIdx.x * blockDim.x + threadIdx.x;   // 64*256
    if (i >= 64 * 256) return;
    int m = i >> 8, k = i & 255;
    dst[i] = (k < 243) ? (_Float16)w[m * 243 + k] : (_Float16)0.0f;
}

// k -> input offset tables (hoists div/mod chains out of the conv hot loops)
// Padding K (k>=243) points at offset 0: A columns there are zero, so the
// loaded value is multiplied by 0 -- no predicate needed in the hot loop.
__global__ void build_ofs_conv1(int* __restrict__ ofs) {
    int k = blockIdx.x * blockDim.x + threadIdx.x;
    if (k >= 256) return;
    if (k < 243) {
        int cin = k / 81, rs = k - cin * 81, r = rs / 9, s = rs - r * 9;
        ofs[k] = cin * 4096 + r * 64 + s;          // x is [3][64][64] per image
    } else {
        ofs[k] = 0;                                // dummy valid offset
    }
}
__global__ void build_ofs_conv2(int* __restrict__ ofs) {
    int k = blockIdx.x * blockDim.x + threadIdx.x;
    if (k >= 5184) return;
    int cin = k / 81, rs = k - cin * 81, r = rs / 9, s = rs - r * 9;
    ofs[k] = cin * 3136 + r * 56 + s;              // h is [64][56][56] per image
}

// ---------------------------------------------------------------------------
// conv1: implicit GEMM, M=64, K=243(pad 256), N=256*56*56. out: f16 NCHW + ReLU
// macro tile 64x64, 8 waves, ping-pong staged B, 1 barrier / K-chunk
// ---------------------------------------------------------------------------
__global__ __launch_bounds__(256) void conv1_wmma(
    const float* __restrict__ x,        // [256][3][64][64]
    const _Float16* __restrict__ A1,    // [64][256]
    const int* __restrict__ ofs,        // [256]
    const float* __restrict__ bias,     // [64]
    _Float16* __restrict__ h)           // [256][64][56][56]
{
    __shared__ _Float16 Bs[2][64][40];  // [buf][col][kk], padded rows (80B)
    const int tid  = threadIdx.x;
    const int lane = tid & 31, wave = tid >> 5;
    const int hi = lane >> 4, ln = lane & 15;
    const int n_block = blockIdx.x * 64;
    const int m_tile  = (wave & 3) * 16;
    const int n_half  = (wave >> 2) * 32;

    const int scol = tid & 63;
    const int skk0 = (tid >> 6) << 3;
    const int sn   = n_block + scol;
    const int sb   = sn / 3136;
    const int srem = sn - sb * 3136;
    const int soh  = srem / 56, sow = srem - soh * 56;
    const float* xb = x + ((size_t)sb * 3) * 4096 + soh * 64 + sow;

    auto stage = [&](int buf, int kc) {
        int4 o0 = *(const int4*)(ofs + kc + skk0);
        int4 o1 = *(const int4*)(ofs + kc + skk0 + 4);
        int o[8] = {o0.x, o0.y, o0.z, o0.w, o1.x, o1.y, o1.z, o1.w};
        h8 v;
        #pragma unroll
        for (int q = 0; q < 8; ++q) v[q] = (_Float16)xb[o[q]];
        *(h8*)&Bs[buf][scol][skk0] = v;
    };

    v8f acc0 = {}, acc1 = {};
    const _Float16* Arow = A1 + (m_tile + ln) * 256;
    stage(0, 0);
    int cur = 0;
    for (int kc = 0; kc < 256; kc += 32) {
        __syncthreads();
        if (kc + 32 < 256) stage(cur ^ 1, kc + 32);
        v16h a  = load_A_frag(Arow + kc, hi);
        v16h b0 = load_B_frag(&Bs[cur][n_half + ln][0], hi);
        v16h b1 = load_B_frag(&Bs[cur][n_half + 16 + ln][0], hi);
        acc0 = __builtin_amdgcn_wmma_f32_16x16x32_f16(false, a, false, b0,
                                                      (short)0, acc0, false, false);
        acc1 = __builtin_amdgcn_wmma_f32_16x16x32_f16(false, a, false, b1,
                                                      (short)0, acc1, false, false);
        cur ^= 1;
    }
    #pragma unroll
    for (int t2 = 0; t2 < 2; ++t2) {
        const v8f& acc = t2 ? acc1 : acc0;
        int n   = n_block + n_half + t2 * 16 + ln;
        int b   = n / 3136;
        int rem = n - b * 3136;
        int oh  = rem / 56, ow = rem - oh * 56;
        #pragma unroll
        for (int r = 0; r < 8; ++r) {
            int m = m_tile + r + hi * 8;
            float v = acc[r] + bias[m];
            v = v > 0.0f ? v : 0.0f;
            h[((b * 64 + m) * 56 + oh) * 56 + ow] = (_Float16)v;
        }
    }
}

// ---------------------------------------------------------------------------
// conv2: implicit GEMM, M=64, K=5184, N=256*24*24, stride 2. out: f32 NCHW
// ---------------------------------------------------------------------------
__global__ __launch_bounds__(256) void conv2_wmma(
    const _Float16* __restrict__ h,     // [256][64][56][56]
    const _Float16* __restrict__ A2,    // [64][5184]
    const int* __restrict__ ofs,        // [5184]
    const float* __restrict__ bias,     // [64]
    float* __restrict__ p)              // [256][64][24][24]
{
    __shared__ _Float16 Bs[2][64][40];
    const int tid  = threadIdx.x;
    const int lane = tid & 31, wave = tid >> 5;
    const int hi = lane >> 4, ln = lane & 15;
    const int n_block = blockIdx.x * 64;
    const int m_tile  = (wave & 3) * 16;
    const int n_half  = (wave >> 2) * 32;

    const int scol = tid & 63;
    const int skk0 = (tid >> 6) << 3;
    const int sn   = n_block + scol;
    const int sb   = sn / 576;
    const int srem = sn - sb * 576;
    const int soh  = srem / 24, sow = srem - soh * 24;
    const _Float16* hb = h + ((size_t)sb * 64) * 3136 + (soh * 2) * 56 + sow * 2;

    auto stage = [&](int buf, int kc) {
        int4 o0 = *(const int4*)(ofs + kc + skk0);
        int4 o1 = *(const int4*)(ofs + kc + skk0 + 4);
        int o[8] = {o0.x, o0.y, o0.z, o0.w, o1.x, o1.y, o1.z, o1.w};
        h8 v;
        #pragma unroll
        for (int q = 0; q < 8; ++q) v[q] = hb[o[q]];
        *(h8*)&Bs[buf][scol][skk0] = v;
    };

    v8f acc0 = {}, acc1 = {};
    const _Float16* Arow = A2 + (m_tile + ln) * 5184;
    stage(0, 0);
    int cur = 0;
    for (int kc = 0; kc < 5184; kc += 32) {
        __syncthreads();
        if (kc + 32 < 5184) stage(cur ^ 1, kc + 32);
        v16h a  = load_A_frag(Arow + kc, hi);
        v16h b0 = load_B_frag(&Bs[cur][n_half + ln][0], hi);
        v16h b1 = load_B_frag(&Bs[cur][n_half + 16 + ln][0], hi);
        acc0 = __builtin_amdgcn_wmma_f32_16x16x32_f16(false, a, false, b0,
                                                      (short)0, acc0, false, false);
        acc1 = __builtin_amdgcn_wmma_f32_16x16x32_f16(false, a, false, b1,
                                                      (short)0, acc1, false, false);
        cur ^= 1;
    }
    #pragma unroll
    for (int t2 = 0; t2 < 2; ++t2) {
        const v8f& acc = t2 ? acc1 : acc0;
        int n   = n_block + n_half + t2 * 16 + ln;
        int b   = n / 576;
        int rem = n - b * 576;
        int oh  = rem / 24, ow = rem - oh * 24;
        #pragma unroll
        for (int r = 0; r < 8; ++r) {
            int m = m_tile + r + hi * 8;
            p[((b * 64 + m) * 24 + oh) * 24 + ow] = acc[r] + bias[m];
        }
    }
}

// ---------------------------------------------------------------------------
// squash primary capsules in place: p viewed as [256*4608][8]
// ---------------------------------------------------------------------------
__global__ void squash_caps(float* __restrict__ p, int ncaps) {
    int i = blockIdx.x * blockDim.x + threadIdx.x;
    if (i >= ncaps) return;
    float* q = p + (size_t)i * 8;
    float4 lo = *reinterpret_cast<float4*>(q);
    float4 hv = *reinterpret_cast<float4*>(q + 4);
    float n2 = lo.x*lo.x + lo.y*lo.y + lo.z*lo.z + lo.w*lo.w
             + hv.x*hv.x + hv.y*hv.y + hv.z*hv.z + hv.w*hv.w;
    float sc = (n2 / (1.0f + n2)) * rsqrtf(n2 + 1e-8f);
    lo.x *= sc; lo.y *= sc; lo.z *= sc; lo.w *= sc;
    hv.x *= sc; hv.y *= sc; hv.z *= sc; hv.w *= sc;
    *reinterpret_cast<float4*>(q)     = lo;
    *reinterpret_cast<float4*>(q + 4) = hv;
}

// ---------------------------------------------------------------------------
// fused dynamic routing: 1 block per batch element, 512 threads (9 caps each)
// ---------------------------------------------------------------------------
__global__ __launch_bounds__(512) void routing_fused(
    const float* __restrict__ u,        // [256][4608][8] (squashed)
    const float* __restrict__ W,        // [4][4608][16][8]
    float* __restrict__ lengths,        // [256][4] -> d_out
    _Float16* __restrict__ z16)         // [256][64]
{
    const int b   = blockIdx.x;
    const int tid = threadIdx.x;
    const int wave = tid >> 5, lane = tid & 31;
    const float* ub = u + (size_t)b * 4608 * 8;

    __shared__ float red[16][4][16];
    __shared__ float vsh[4][16];
    __shared__ float scl[4];
    __shared__ float len_sh[4];
    __shared__ int   omax_sh;

    float blog[9][4];
    #pragma unroll
    for (int j = 0; j < 9; ++j)
        #pragma unroll
        for (int o = 0; o < 4; ++o) blog[j][o] = 0.0f;

    for (int it = 0; it < 3; ++it) {
        float vacc[4][16];
        #pragma unroll
        for (int o = 0; o < 4; ++o)
            #pragma unroll
            for (int d = 0; d < 16; ++d) vacc[o][d] = 0.0f;

        for (int j = 0; j < 9; ++j) {
            int i = tid + j * 512;
            float mx = blog[j][0];
            #pragma unroll
            for (int o = 1; o < 4; ++o) mx = fmaxf(mx, blog[j][o]);
            float e[4], s = 0.0f;
            #pragma unroll
            for (int o = 0; o < 4; ++o) { e[o] = __expf(blog[j][o] - mx); s += e[o]; }
            float inv = 1.0f / s;
            float uk[8];
            #pragma unroll
            for (int k = 0; k < 8; ++k) uk[k] = ub[i * 8 + k];
            const float* Wi = W + (size_t)i * 128;
            #pragma unroll
            for (int o = 0; o < 4; ++o) {
                float c = e[o] * inv;
                const float* Wo = Wi + (size_t)o * 4608 * 128;
                #pragma unroll
                for (int d = 0; d < 16; ++d) {
                    const float* wp = Wo + d * 8;
                    float xh = wp[0]*uk[0] + wp[1]*uk[1] + wp[2]*uk[2] + wp[3]*uk[3]
                             + wp[4]*uk[4] + wp[5]*uk[5] + wp[6]*uk[6] + wp[7]*uk[7];
                    vacc[o][d] += c * xh;
                }
            }
        }
        #pragma unroll
        for (int o = 0; o < 4; ++o)
            #pragma unroll
            for (int d = 0; d < 16; ++d)
                for (int off = 16; off >= 1; off >>= 1)
                    vacc[o][d] += __shfl_down(vacc[o][d], off, 32);
        if (lane == 0)
            for (int o = 0; o < 4; ++o)
                for (int d = 0; d < 16; ++d) red[wave][o][d] = vacc[o][d];
        __syncthreads();
        if (tid < 64) {
            int o = tid >> 4, d = tid & 15;
            float s = 0.0f;
            #pragma unroll
            for (int w = 0; w < 16; ++w) s += red[w][o][d];
            vsh[o][d] = s;
        }
        __syncthreads();
        if (tid < 4) {
            float n2 = 0.0f;
            #pragma unroll
            for (int d = 0; d < 16; ++d) n2 += vsh[tid][d] * vsh[tid][d];
            scl[tid] = (n2 / (1.0f + n2)) * rsqrtf(n2 + 1e-8f);
        }
        __syncthreads();

        if (it < 2) {
            float vr[4][16];
            #pragma unroll
            for (int o = 0; o < 4; ++o)
                #pragma unroll
                for (int d = 0; d < 16; ++d) vr[o][d] = vsh[o][d] * scl[o];
            for (int j = 0; j < 9; ++j) {
                int i = tid + j * 512;
                float uk[8];
                #pragma unroll
                for (int k = 0; k < 8; ++k) uk[k] = ub[i * 8 + k];
                const float* Wi = W + (size_t)i * 128;
                #pragma unroll
                for (int o = 0; o < 4; ++o) {
                    const float* Wo = Wi + (size_t)o * 4608 * 128;
                    float dot = 0.0f;
                    #pragma unroll
                    for (int d = 0; d < 16; ++d) {
                        const float* wp = Wo + d * 8;
                        float xh = wp[0]*uk[0] + wp[1]*uk[1] + wp[2]*uk[2] + wp[3]*uk[3]
                                 + wp[4]*uk[4] + wp[5]*uk[5] + wp[6]*uk[6] + wp[7]*uk[7];
                        dot += vr[o][d] * xh;
                    }
                    blog[j][o] += dot;
                }
            }
            __syncthreads();
        } else {
            if (tid < 4) {
                float n2 = 0.0f;
                #pragma unroll
                for (int d = 0; d < 16; ++d) {
                    float sv = vsh[tid][d] * scl[tid];
                    n2 += sv * sv;
                }
                float len = sqrtf(n2);
                len_sh[tid] = len;
                lengths[b * 4 + tid] = len;
            }
            __syncthreads();
            if (tid == 0) {
                int am = 0; float best = len_sh[0];
                for (int o = 1; o < 4; ++o)
                    if (len_sh[o] > best) { best = len_sh[o]; am = o; }
                omax_sh = am;
            }
            __syncthreads();
            if (tid < 64) {
                int o = tid >> 4, d = tid & 15;
                float z = (o == omax_sh) ? vsh[o][d] * scl[o] : 0.0f;
                z16[b * 64 + tid] = (_Float16)z;
            }
        }
    }
}

// ---------------------------------------------------------------------------
// generic WMMA GEMM: C[M][N] = act(A[M][K] * B[K][N] + bias[N])
// act: 0 = relu (f16 out), 1 = sigmoid (f32 out). M%64==0, N%64==0, K%32==0.
// ---------------------------------------------------------------------------
__global__ __launch_bounds__(256) void gemm_wmma_f16(
    const _Float16* __restrict__ A, const _Float16* __restrict__ B,
    const float* __restrict__ bias,
    _Float16* __restrict__ Cf16, float* __restrict__ Cf32,
    int M, int N, int K, int act)
{
    __shared__ _Float16 Bs[2][64][40];
    const int tid  = threadIdx.x;
    const int lane = tid & 31, wave = tid >> 5;
    const int hi = lane >> 4, ln = lane & 15;
    const int n_block = blockIdx.x * 64;
    const int m_block = blockIdx.y * 64;
    const int m_tile  = (wave & 3) * 16;
    const int n_half  = (wave >> 2) * 32;

    const int scol = tid & 63;
    const int skk0 = (tid >> 6) << 3;
    const _Float16* Bsrc = B + (size_t)skk0 * N + n_block + scol;

    auto stage = [&](int buf, int kc) {
        const _Float16* bp = Bsrc + (size_t)kc * N;
        h8 v;
        #pragma unroll
        for (int q = 0; q < 8; ++q) v[q] = bp[(size_t)q * N];
        *(h8*)&Bs[buf][scol][skk0] = v;
    };

    v8f acc0 = {}, acc1 = {};
    const _Float16* Arow = A + (size_t)(m_block + m_tile + ln) * K;
    stage(0, 0);
    int cur = 0;
    for (int kc = 0; kc < K; kc += 32) {
        __syncthreads();
        if (kc + 32 < K) stage(cur ^ 1, kc + 32);
        v16h a  = load_A_frag(Arow + kc, hi);
        v16h b0 = load_B_frag(&Bs[cur][n_half + ln][0], hi);
        v16h b1 = load_B_frag(&Bs[cur][n_half + 16 + ln][0], hi);
        acc0 = __builtin_amdgcn_wmma_f32_16x16x32_f16(false, a, false, b0,
                                                      (short)0, acc0, false, false);
        acc1 = __builtin_amdgcn_wmma_f32_16x16x32_f16(false, a, false, b1,
                                                      (short)0, acc1, false, false);
        cur ^= 1;
    }
    #pragma unroll
    for (int t2 = 0; t2 < 2; ++t2) {
        const v8f& acc = t2 ? acc1 : acc0;
        int n = n_block + n_half + t2 * 16 + ln;
        float bia = bias[n];
        #pragma unroll
        for (int r = 0; r < 8; ++r) {
            int m = m_block + m_tile + r + hi * 8;
            float v = acc[r] + bia;
            if (act == 0) {
                v = v > 0.0f ? v : 0.0f;
                Cf16[(size_t)m * N + n] = (_Float16)v;
            } else {
                Cf32[(size_t)m * N + n] = 1.0f / (1.0f + __expf(-v));
            }
        }
    }
}

// ---------------------------------------------------------------------------
// launch
// ---------------------------------------------------------------------------
extern "C" void kernel_launch(void* const* d_in, const int* in_sizes, int n_in,
                              void* d_out, int out_size, void* d_ws, size_t ws_size,
                              hipStream_t stream) {
    const float* x       = (const float*)d_in[0];
    const float* conv1_w = (const float*)d_in[1];
    const float* conv1_b = (const float*)d_in[2];
    const float* conv2_w = (const float*)d_in[3];
    const float* conv2_b = (const float*)d_in[4];
    const float* digit_W = (const float*)d_in[5];
    const float* dec1_w  = (const float*)d_in[6];
    const float* dec1_b  = (const float*)d_in[7];
    const float* dec2_w  = (const float*)d_in[8];
    const float* dec2_b  = (const float*)d_in[9];
    const float* dec3_w  = (const float*)d_in[10];
    const float* dec3_b  = (const float*)d_in[11];

    float* lengths = (float*)d_out;              // [256][4]
    float* recon   = (float*)d_out + 1024;       // [256][12288]

    char*  base = (char*)d_ws;
    size_t off  = 0;
    auto carve = [&](size_t bytes) -> void* {
        void* p = base + off;
        off += (bytes + 255) & ~(size_t)255;
        return p;
    };
    _Float16* A1h  = (_Float16*)carve(64 * 256 * 2);
    _Float16* A2h  = (_Float16*)carve(64 * 5184 * 2);
    int*      ofs1 = (int*)carve(256 * 4);
    int*      ofs2 = (int*)carve(5184 * 4);
    _Float16* W1h  = (_Float16*)carve(64 * 512 * 2);
    _Float16* W2h  = (_Float16*)carve(512 * 1024 * 2);
    _Float16* W3h  = (_Float16*)carve((size_t)1024 * 12288 * 2);
    _Float16* h16  = (_Float16*)carve((size_t)256 * 64 * 56 * 56 * 2);
    float*    pbuf = (float*)carve((size_t)256 * 64 * 24 * 24 * 4);
    _Float16* z16  = (_Float16*)carve(256 * 64 * 2);
    _Float16* a1h  = (_Float16*)carve(256 * 512 * 2);
    _Float16* a2h  = (_Float16*)carve(256 * 1024 * 2);
    (void)ws_size; (void)in_sizes; (void)n_in; (void)out_size;

    build_A1<<<(64 * 256 + 255) / 256, 256, 0, stream>>>(conv1_w, A1h);
    build_ofs_conv1<<<1, 256, 0, stream>>>(ofs1);
    build_ofs_conv2<<<(5184 + 255) / 256, 256, 0, stream>>>(ofs2);
    cvt_f32_to_f16<<<(64 * 5184 + 255) / 256, 256, 0, stream>>>(conv2_w, A2h, 64 * 5184);
    cvt_f32_to_f16<<<(64 * 512 + 255) / 256, 256, 0, stream>>>(dec1_w, W1h, 64 * 512);
    cvt_f32_to_f16<<<(512 * 1024 + 255) / 256, 256, 0, stream>>>(dec2_w, W2h, 512 * 1024);
    cvt_f32_to_f16<<<(1024 * 12288 + 255) / 256, 256, 0, stream>>>(dec3_w, W3h, 1024 * 12288);

    conv1_wmma<<<802816 / 64, 256, 0, stream>>>(x, A1h, ofs1, conv1_b, h16);
    conv2_wmma<<<147456 / 64, 256, 0, stream>>>(h16, A2h, ofs2, conv2_b, pbuf);
    squash_caps<<<(256 * 4608 + 255) / 256, 256, 0, stream>>>(pbuf, 256 * 4608);
    routing_fused<<<256, 512, 0, stream>>>(pbuf, digit_W, lengths, z16);

    gemm_wmma_f16<<<dim3(512 / 64, 256 / 64), 256, 0, stream>>>(
        z16, W1h, dec1_b, a1h, nullptr, 256, 512, 64, 0);
    gemm_wmma_f16<<<dim3(1024 / 64, 256 / 64), 256, 0, stream>>>(
        a1h, W2h, dec2_b, a2h, nullptr, 256, 1024, 512, 0);
    gemm_wmma_f16<<<dim3(12288 / 64, 256 / 64), 256, 0, stream>>>(
        a2h, W3h, dec3_b, nullptr, recon, 256, 12288, 1024, 1);
}
